// TransformerLayer_85048942395852
// MI455X (gfx1250) — matrix operand order repaired and checked by reading, not compiled, not run on hardware
//
#include <hip/hip_runtime.h>
#include <stdint.h>

static constexpr int kN   = 16384;
static constexpr int kE   = 8192;
static constexpr int kC   = 256;
static constexpr int kNNZ = 262144;

typedef float        v8f   __attribute__((ext_vector_type(8)));
typedef __bf16       v16bf __attribute__((ext_vector_type(16)));
typedef __bf16       v8bf  __attribute__((ext_vector_type(8)));
typedef unsigned int v4u   __attribute__((ext_vector_type(4)));
typedef int          v8i   __attribute__((ext_vector_type(8)));
typedef int          v4i   __attribute__((ext_vector_type(4)));

#if defined(__HIP_DEVICE_COMPILE__) && __has_builtin(__builtin_amdgcn_tensor_load_to_lds) && __has_builtin(__builtin_amdgcn_s_wait_tensorcnt)
#define USE_TDM 1
#else
#define USE_TDM 0
#endif

// ---------------------------------------------------------------- helpers

__device__ __forceinline__ v8f wmma_bf16(v16bf a, v16bf b, v8f c) {
  // v_wmma_f32_16x16x32_bf16  (8-arg probe-confirmed signature)
  return __builtin_amdgcn_wmma_f32_16x16x32_bf16(false, a, false, b, (short)0, c,
                                                 false, false);
}

// A-fragment style load: two separated 16-byte halfword chunks -> v16bf
__device__ __forceinline__ v16bf load_frag2(const __bf16* p0, const __bf16* p1) {
  union { v16bf v; v8bf h[2]; } u;
  u.h[0] = *(const v8bf*)p0;
  u.h[1] = *(const v8bf*)p1;
  return u.v;
}

#if USE_TDM
// 2D TENSOR_LOAD_TO_LDS. Element size fixed to 2 bytes (bf16).
// tensor_d0 = elements per row, tensor_d1 = rows, tile_d0 = tile width
// (contiguous), tile_d1 = tile rows, stride0 = row stride in elements.
// Toolchain uses the 6-arg builtin: (v4u g0, v8i g1, v4i g2, v4i g3, v8i g4, i32 cpol).
__device__ __forceinline__ void tdm_load_2d(const void* gaddr, unsigned lds_off,
                                            unsigned tensor_d0, unsigned tensor_d1,
                                            unsigned tile_d0, unsigned tile_d1,
                                            unsigned long long stride0) {
  unsigned long long ga = (unsigned long long)(uintptr_t)gaddr;
  v4u g0;
  g0[0] = 1u;                                           // count=1, user descriptor
  g0[1] = lds_off;                                      // lds_addr
  g0[2] = (unsigned)(ga & 0xFFFFFFFFu);                 // global_addr[31:0]
  g0[3] = (unsigned)((ga >> 32) & 0x1FFFFFFu) | (2u << 30); // ga[56:32] | type=2
  v8i g1;
  g1[0] = (int)(1u << 16);                              // data_size=1 (2 bytes)
  g1[1] = (int)((tensor_d0 & 0xFFFFu) << 16);           // tensor_dim0[15:0]
  g1[2] = (int)(((tensor_d0 >> 16) & 0xFFFFu) | ((tensor_d1 & 0xFFFFu) << 16));
  g1[3] = (int)(((tensor_d1 >> 16) & 0xFFFFu) | ((tile_d0 & 0xFFFFu) << 16));
  g1[4] = (int)(tile_d1 & 0xFFFFu);                     // tile_dim1, tile_dim2=0
  g1[5] = (int)(unsigned)(stride0 & 0xFFFFFFFFull);     // tensor_dim0_stride[31:0]
  g1[6] = (int)(unsigned)((stride0 >> 32) & 0xFFFFull); // stride0[47:32]
  g1[7] = 0;
  v4i gz4; gz4[0] = gz4[1] = gz4[2] = gz4[3] = 0;
  v8i gz8;
  gz8[0] = gz8[1] = gz8[2] = gz8[3] = 0;
  gz8[4] = gz8[5] = gz8[6] = gz8[7] = 0;
  __builtin_amdgcn_tensor_load_to_lds(g0, g1, gz4, gz4, gz8, 0);
}
#endif

// ---------------------------------------------------------------- small kernels

__global__ void k_cvt_x(const float* __restrict__ X, __bf16* __restrict__ Xb) {
  int tid = blockIdx.x * 256 + threadIdx.x;
  int i = tid * 4;
  if (i >= kN * kC) return;
  float4 x = *(const float4*)(X + i);
  Xb[i + 0] = (__bf16)x.x; Xb[i + 1] = (__bf16)x.y;
  Xb[i + 2] = (__bf16)x.z; Xb[i + 3] = (__bf16)x.w;
}

__global__ void k_cvt_w(const float* __restrict__ Wq, const float* __restrict__ Wk,
                        const float* __restrict__ Wv, const float* __restrict__ Wke,
                        __bf16* __restrict__ WqT, __bf16* __restrict__ WkT,
                        __bf16* __restrict__ WvT, __bf16* __restrict__ WkeT) {
  int tid = blockIdx.x * 256 + threadIdx.x;
  if (tid >= 4 * kC * kC) return;
  int which = tid >> 16;
  int rem = tid & 65535;                 // rem = o*256 + c
  int o = rem >> 8, c = rem & 255;
  const float* W = which == 0 ? Wq : which == 1 ? Wk : which == 2 ? Wv : Wke;
  __bf16* T = which == 0 ? WqT : which == 1 ? WkT : which == 2 ? WvT : WkeT;
  T[rem] = (__bf16)W[c * kC + o];        // WT[o][c] = W[c][o]
}

__global__ void k_zero(float* __restrict__ XeSum, float* __restrict__ cnt) {
  int tid = blockIdx.x * 256 + threadIdx.x;
  if (tid < kE * kC) XeSum[tid] = 0.f;
  if (tid < kE) cnt[tid] = 0.f;
}

__global__ void k_count(const int* __restrict__ e_ids, float* __restrict__ cnt) {
  int tid = blockIdx.x * 256 + threadIdx.x;
  if (tid < kNNZ) atomicAdd(&cnt[e_ids[tid]], 1.f);
}

__global__ void k_scatter(const float* __restrict__ X, const int* __restrict__ v_ids,
                          const int* __restrict__ e_ids, float* __restrict__ XeSum) {
  long long t = (long long)blockIdx.x * 256 + threadIdx.x;
  int p = (int)(t >> 6);
  int c4 = ((int)t & 63) * 4;
  if (p >= kNNZ) return;
  int vtx = v_ids[p], e = e_ids[p];
  float4 x = *(const float4*)(X + (size_t)vtx * kC + c4);
  float* dst = XeSum + (size_t)e * kC + c4;
  atomicAdd(dst + 0, x.x); atomicAdd(dst + 1, x.y);
  atomicAdd(dst + 2, x.z); atomicAdd(dst + 3, x.w);
}

__global__ void k_finalize(const float* __restrict__ XeSum, const float* __restrict__ cnt,
                           __bf16* __restrict__ XeB, __bf16* __restrict__ XeT) {
  int tid = blockIdx.x * 256 + threadIdx.x;
  if (tid >= kE * kC) return;
  int e = tid >> 8, c = tid & 255;
  float v = XeSum[tid] / fmaxf(cnt[e], 1.f);
  XeB[tid] = (__bf16)v;                  // row-major [E,C]
  XeT[(size_t)c * kE + e] = (__bf16)v;   // transposed [C,E]
}

// ---------------------------------------------------------------- WMMA GEMM: out = A @ W + b
// A: [M,256] bf16 row-major.  WT: [256(out),256(in)] bf16.  Wave: 16 rows x 64 cols.

__global__ void __launch_bounds__(128) k_gemm_bias(
    const __bf16* __restrict__ A, const __bf16* __restrict__ WT,
    const float* __restrict__ bias, float* __restrict__ outF,
    __bf16* __restrict__ outB, int M) {
  const int lane = threadIdx.x & 31;
  const int w = threadIdx.x >> 5;
  const int gid = blockIdx.x * 4 + w;
  const int rowTiles = M >> 4;
  if (gid >= rowTiles * 4) return;
  const int rt = gid >> 2, ct = gid & 3;
  const int r0 = rt * 16, c0 = ct * 64;
  const int m16 = lane & 15, half = lane >> 4;
  const int kb0 = half * 16;   // A-frag halfword base  (K 0..15 / 16..31)
  const int kbB = half * 16;   // B-frag halfword base  (K 0..15 / 16..31)

  v8f acc[4];
  #pragma unroll
  for (int j = 0; j < 4; ++j)
    #pragma unroll
    for (int r = 0; r < 8; ++r) acc[j][r] = 0.f;

  #pragma unroll
  for (int kc = 0; kc < 8; ++kc) {
    const __bf16* ap = A + (size_t)(r0 + m16) * kC + kc * 32 + kb0;
    v16bf a = load_frag2(ap, ap + 8);
    #pragma unroll
    for (int j = 0; j < 4; ++j) {
      const __bf16* bp = WT + (size_t)(c0 + j * 16 + m16) * kC + kc * 32 + kbB;
      acc[j] = wmma_bf16(a, *(const v16bf*)bp, acc[j]);
    }
  }
  #pragma unroll
  for (int j = 0; j < 4; ++j) {
    const int col = c0 + j * 16 + m16;
    const float b = bias[col];
    #pragma unroll
    for (int r = 0; r < 8; ++r) {
      const int row = r0 + half * 8 + r;
      float v = acc[j][r] + b;
      if (outF) outF[(size_t)row * kC + col] = v;
      if (outB) outB[(size_t)row * kC + col] = (__bf16)v;
    }
  }
}

// ---------------------------------------------------------------- per-node 8x8 head attention (tiny, VALU)

__global__ void __launch_bounds__(256) k_node_attn(
    const float* __restrict__ Qf, const float* __restrict__ Kf,
    const float* __restrict__ Vf, float* __restrict__ Xnode) {
  __shared__ float sQ[8][256];
  __shared__ float sK[8][256];
  __shared__ float sV[8][256];
  __shared__ float sS[8][64];
  const int w = threadIdx.x >> 5;
  const int lane = threadIdx.x & 31;
  const int node = blockIdx.x * 8 + w;
  const float scale = 0.17677669529663687f; // 1/sqrt(32)
  const float* q = Qf + (size_t)node * kC;
  const float* k = Kf + (size_t)node * kC;
  const float* v = Vf + (size_t)node * kC;
  #pragma unroll
  for (int i = 0; i < 8; ++i) {
    sQ[w][lane + 32 * i] = q[lane + 32 * i];
    sK[w][lane + 32 * i] = k[lane + 32 * i];
    sV[w][lane + 32 * i] = v[lane + 32 * i];
  }
  #pragma unroll
  for (int e = lane; e < 64; e += 32) {
    int h = e >> 3, g = e & 7;
    float acc = 0.f;
    #pragma unroll
    for (int d = 0; d < 32; ++d) acc += sQ[w][h * 32 + d] * sK[w][g * 32 + d];
    sS[w][e] = acc * scale;
  }
  if (lane < 8) {
    float mx = -3.0e38f;
    #pragma unroll
    for (int g = 0; g < 8; ++g) mx = fmaxf(mx, sS[w][lane * 8 + g]);
    float s = 0.f;
    #pragma unroll
    for (int g = 0; g < 8; ++g) {
      float p = __expf(sS[w][lane * 8 + g] - mx);
      sS[w][lane * 8 + g] = p; s += p;
    }
    float inv = 1.f / s;
    #pragma unroll
    for (int g = 0; g < 8; ++g) sS[w][lane * 8 + g] *= inv;
  }
  #pragma unroll
  for (int j = 0; j < 8; ++j) {
    int c = lane + 32 * j;
    int h = c >> 5, d = c & 31;
    float acc = 0.f;
    #pragma unroll
    for (int g = 0; g < 8; ++g) acc += sS[w][h * 8 + g] * sV[w][g * 32 + d];
    Xnode[(size_t)node * kC + c] = acc;
  }
}

// ---------------------------------------------------------------- flash attention over hyperedges
// out = relu(Xnode + softmax(Q @ Ke^T * s) @ Xe). 4 waves/block, 16 rows/wave.

__global__ void __launch_bounds__(128) k_edge_attn(
    const __bf16* __restrict__ Qb, const __bf16* __restrict__ Keb,
    const __bf16* __restrict__ XeT, const float* __restrict__ Xnode,
    float* __restrict__ out) {
  __shared__ __bf16 sKe[2][32 * 256];   // key-chunk rows, row-major
  __shared__ __bf16 sXe[2][256 * 32];   // XeT rows (channels) x 32 keys
  __shared__ __bf16 sQ[4][16 * 256];    // per-wave Q tile
  __shared__ __bf16 sP[4][16 * 32];     // per-wave probabilities (C->A relayout)

  const int lane = threadIdx.x & 31;
  const int w = threadIdx.x >> 5;
  const int q0 = blockIdx.x * 64 + w * 16;
  const int m16 = lane & 15;
  const int half = lane >> 4;
  const int rowBase = half * 8;
  const int kb0 = half * 16;
  const int kbB = half * 16;
  const float scale = 0.17677669529663687f; // 1/sqrt(32)
  const int NCH = kE / 32;                  // 256 key chunks

  { // stage this wave's Q tile into LDS (16x256 bf16 = 8KB)
    const v8bf* src = (const v8bf*)(Qb + (size_t)q0 * kC);
    v8bf* dst = (v8bf*)sQ[w];
    #pragma unroll
    for (int i = 0; i < 16; ++i) dst[lane + 32 * i] = src[lane + 32 * i];
  }

  v8f o[16];
  #pragma unroll
  for (int t = 0; t < 16; ++t)
    #pragma unroll
    for (int r = 0; r < 8; ++r) o[t][r] = 0.f;
  float m[8], l[8];
  #pragma unroll
  for (int r = 0; r < 8; ++r) { m[r] = -3.0e38f; l[r] = 0.f; }

#if USE_TDM
  if (w == 0) { // prefetch chunk 0 via the Tensor Data Mover
    tdm_load_2d(Keb, (unsigned)(uintptr_t)&sKe[0][0], kC, kE, kC, 32, kC);
    tdm_load_2d(XeT, (unsigned)(uintptr_t)&sXe[0][0], kE, kC, 32, kC, kE);
  }
#endif

  for (int it = 0; it < NCH; ++it) {
#if USE_TDM
    const int buf = it & 1;
    if (w == 0 && it + 1 < NCH) { // issue next chunk into the other buffer
      const int nb = buf ^ 1;
      tdm_load_2d(Keb + (size_t)(it + 1) * 32 * kC,
                  (unsigned)(uintptr_t)&sKe[nb][0], kC, kE, kC, 32, kC);
      tdm_load_2d(XeT + (size_t)(it + 1) * 32,
                  (unsigned)(uintptr_t)&sXe[nb][0], kE, kC, 32, kC, kE);
    }
    if (w == 0) { // TDM is in-order per wave: cnt<=2 means current chunk landed
      if (it + 1 < NCH) __builtin_amdgcn_s_wait_tensorcnt(2);
      else              __builtin_amdgcn_s_wait_tensorcnt(0);
    }
    __syncthreads();
#else
    const int buf = 0;
    __syncthreads();
    { // cooperative synchronous staging fallback
      const int tid = threadIdx.x;
      const v8bf* srcK = (const v8bf*)(Keb + (size_t)it * 32 * kC);
      v8bf* dstK = (v8bf*)sKe[0];
      #pragma unroll
      for (int i = 0; i < 8; ++i) dstK[tid + 128 * i] = srcK[tid + 128 * i];
      v8bf* dstX = (v8bf*)sXe[0];
      #pragma unroll
      for (int i = 0; i < 8; ++i) {
        int qq = tid + 128 * i;  // 1024 chunks of 16B
        int row = qq >> 2, part = qq & 3;
        dstX[qq] = *(const v8bf*)(XeT + (size_t)row * kE + (size_t)it * 32 + part * 8);
      }
    }
    __syncthreads();
#endif

    // ---- S = Q @ Ke^T for 32 keys (two 16x16 tiles, K=256)
    v8f s0, s1;
    #pragma unroll
    for (int r = 0; r < 8; ++r) { s0[r] = 0.f; s1[r] = 0.f; }
    #pragma unroll
    for (int kc = 0; kc < 8; ++kc) {
      v16bf a = load_frag2(&sQ[w][m16 * 256 + kc * 32 + kb0],
                           &sQ[w][m16 * 256 + kc * 32 + kb0 + 8]);
      v16bf b0 = *(const v16bf*)&sKe[buf][m16 * 256 + kc * 32 + kbB];
      v16bf b1 = *(const v16bf*)&sKe[buf][(m16 + 16) * 256 + kc * 32 + kbB];
      s0 = wmma_bf16(a, b0, s0);
      s1 = wmma_bf16(a, b1, s1);
    }

    // ---- online softmax (rows striped over VGPR index, keys over 16-lane halves)
    float alpha[8];
    #pragma unroll
    for (int r = 0; r < 8; ++r) {
      float a0 = s0[r] * scale;
      float a1 = s1[r] * scale;
      float mx = fmaxf(a0, a1);
      #pragma unroll
      for (int d = 8; d >= 1; d >>= 1) mx = fmaxf(mx, __shfl_xor(mx, d, 32));
      const float mn = fmaxf(m[r], mx);
      const float p0 = __expf(a0 - mn);
      const float p1 = __expf(a1 - mn);
      s0[r] = p0; s1[r] = p1;
      float sum = p0 + p1;
      #pragma unroll
      for (int d = 8; d >= 1; d >>= 1) sum += __shfl_xor(sum, d, 32);
      alpha[r] = __expf(m[r] - mn);
      l[r] = l[r] * alpha[r] + sum;
      m[r] = mn;
    }
    #pragma unroll
    for (int t = 0; t < 16; ++t)
      #pragma unroll
      for (int r = 0; r < 8; ++r) o[t][r] *= alpha[r];

    // ---- relayout P (C-layout -> A-layout) through per-wave LDS, as bf16
    #pragma unroll
    for (int r = 0; r < 8; ++r) {
      sP[w][(rowBase + r) * 32 + m16]      = (__bf16)s0[r];
      sP[w][(rowBase + r) * 32 + 16 + m16] = (__bf16)s1[r];
    }
    v16bf p = load_frag2(&sP[w][m16 * 32 + kb0], &sP[w][m16 * 32 + kb0 + 8]);

    // ---- O += P @ Xe_chunk  (16 column tiles, K=32)
    #pragma unroll
    for (int t = 0; t < 16; ++t) {
      v16bf b = *(const v16bf*)&sXe[buf][(t * 16 + m16) * 32 + kbB];
      o[t] = wmma_bf16(p, b, o[t]);
    }

#if USE_TDM
    __syncthreads(); // all readers done before TDM overwrites this buffer
#endif
  }

  // ---- epilogue: normalize, add node-attention, ReLU
  #pragma unroll
  for (int t = 0; t < 16; ++t) {
    const int col = t * 16 + m16;
    #pragma unroll
    for (int r = 0; r < 8; ++r) {
      const int row = q0 + rowBase + r;
      const float vh = o[t][r] / l[r];
      const float xn = Xnode[(size_t)row * kC + col];
      out[(size_t)row * kC + col] = fmaxf(vh + xn, 0.f);
    }
  }
}

// ---------------------------------------------------------------- launcher

extern "C" void kernel_launch(void* const* d_in, const int* in_sizes, int n_in,
                              void* d_out, int out_size, void* d_ws, size_t ws_size,
                              hipStream_t stream) {
  (void)in_sizes; (void)n_in; (void)out_size; (void)ws_size;
  const float* X    = (const float*)d_in[0];
  const int* v_ids  = (const int*)d_in[1];
  const int* e_ids  = (const int*)d_in[2];
  const float* Wq   = (const float*)d_in[3];
  const float* bq   = (const float*)d_in[4];
  const float* Wk   = (const float*)d_in[5];
  const float* bk   = (const float*)d_in[6];
  const float* Wv   = (const float*)d_in[7];
  const float* bv   = (const float*)d_in[8];
  const float* Wke  = (const float*)d_in[9];
  const float* bke  = (const float*)d_in[10];
  float* out = (float*)d_out;

  char* base = (char*)d_ws;
  size_t off = 0;
  auto alloc = [&](size_t bytes) -> void* {
    void* p = base + off;
    off += (bytes + 255) & ~(size_t)255;
    return p;
  };
  float*  Qf    = (float*)alloc((size_t)kN * kC * 4);
  float*  Kf    = (float*)alloc((size_t)kN * kC * 4);
  float*  Vf    = (float*)alloc((size_t)kN * kC * 4);
  float*  Xnode = (float*)alloc((size_t)kN * kC * 4);
  float*  XeSum = (float*)alloc((size_t)kE * kC * 4);
  float*  cnt   = (float*)alloc((size_t)kE * 4);
  __bf16* Xb    = (__bf16*)alloc((size_t)kN * kC * 2);
  __bf16* Qb    = (__bf16*)alloc((size_t)kN * kC * 2);
  __bf16* XeB   = (__bf16*)alloc((size_t)kE * kC * 2);
  __bf16* XeT   = (__bf16*)alloc((size_t)kE * kC * 2);
  __bf16* Keb   = (__bf16*)alloc((size_t)kE * kC * 2);
  __bf16* WqT   = (__bf16*)alloc((size_t)kC * kC * 2);
  __bf16* WkT   = (__bf16*)alloc((size_t)kC * kC * 2);
  __bf16* WvT   = (__bf16*)alloc((size_t)kC * kC * 2);
  __bf16* WkeT  = (__bf16*)alloc((size_t)kC * kC * 2);

  // conversions
  k_cvt_x<<<(kN * kC / 4 + 255) / 256, 256, 0, stream>>>(X, Xb);
  k_cvt_w<<<(4 * kC * kC + 255) / 256, 256, 0, stream>>>(Wq, Wk, Wv, Wke,
                                                         WqT, WkT, WvT, WkeT);
  // projections (WMMA bf16)
  k_gemm_bias<<<(kN / 16 * 4) / 4, 128, 0, stream>>>(Xb, WqT, bq, Qf, Qb, kN);
  k_gemm_bias<<<(kN / 16 * 4) / 4, 128, 0, stream>>>(Xb, WkT, bk, Kf, nullptr, kN);
  k_gemm_bias<<<(kN / 16 * 4) / 4, 128, 0, stream>>>(Xb, WvT, bv, Vf, nullptr, kN);
  // segment mean -> Xe (row-major bf16) and XeT (transposed bf16)
  k_zero<<<(kE * kC + 255) / 256, 256, 0, stream>>>(XeSum, cnt);
  k_count<<<(kNNZ + 255) / 256, 256, 0, stream>>>(e_ids, cnt);
  k_scatter<<<(kNNZ * 64) / 256, 256, 0, stream>>>(X, v_ids, e_ids, XeSum);
  k_finalize<<<(kE * kC + 255) / 256, 256, 0, stream>>>(XeSum, cnt, XeB, XeT);
  // Ke = Xe @ Wke + bke (WMMA bf16)
  k_gemm_bias<<<(kE / 16 * 4) / 4, 128, 0, stream>>>(XeB, WkeT, bke, nullptr, Keb, kE);
  // tiny per-node head attention
  k_node_attn<<<kN / 8, 256, 0, stream>>>(Qf, Kf, Vf, Xnode);
  // dominant flash attention over hyperedges + fuse add & relu
  k_edge_attn<<<kN / 64, 128, 0, stream>>>(Qb, Keb, XeT, Xnode, out);
}
